// MLSMPO_69930657513826
// MI455X (gfx1250) — compile-verified, matrix operand order checked by reading, not run on hardware
//
#include <hip/hip_runtime.h>

typedef _Float16 half_t;
typedef __attribute__((ext_vector_type(16))) _Float16 v16h;
typedef __attribute__((ext_vector_type(8)))  float    v8f;

constexpr int BDIM = 256;   // 8 wave32 waves
constexpr int ENVS = 68;    // env row stride (floats), padded: bank step 4
constexpr int BTS  = 136;   // Bt row stride (halfs),  padded: bank step 4

// ---------------------------------------------------------------------------
// Cooperative fills of the f16 weight slab Bt[n][k] (K=128=(d|e,p), N=64).
// left  : GEMM is  lv(B x (d,p)) @ W[(d,p), e]   -> new l
// right : GEMM is  rv(B x (e,p)) @ W[(e,p), d]   -> new r
// Wmid layout [d][e][p] (p innermost). Wout layout [d][e][p][f], f summed
// (ones-closure of the output leg) at output sites.
// ---------------------------------------------------------------------------
__device__ __forceinline__ void fill_Bt_mid(half_t* Bt, const float* __restrict__ W,
                                            bool right) {
  for (int q = threadIdx.x; q < 8192; q += BDIM) {
    const int k = q >> 6, nc = q & 63;
    const float w = right ? W[nc * 128 + (k >> 1) * 2 + (k & 1)]
                          : W[(k >> 1) * 128 + nc * 2 + (k & 1)];
    Bt[nc * BTS + k] = (half_t)w;
  }
}

__device__ __forceinline__ void fill_Bt_out(half_t* Bt, const float* __restrict__ W,
                                            bool right) {
  for (int q = threadIdx.x; q < 8192; q += BDIM) {
    const int k = q >> 6, nc = q & 63;
    const float* p = right ? (W + nc * 1152 + (k >> 1) * 18 + (k & 1) * 9)
                           : (W + (k >> 1) * 1152 + nc * 18 + (k & 1) * 9);
    float w = 0.f;
#pragma unroll
    for (int f = 0; f < 9; ++f) w += p[f];
    Bt[nc * BTS + k] = (half_t)w;
  }
}

// ---------------------------------------------------------------------------
// One scan step: env(128x64, f32 in LDS)  <-  [residual? env : 0] +
//                (env .* v) (128x128) @ Bt (128x64), via v_wmma_f32_16x16x32_f16.
// A-fragments are built in-registers from the fp32 env (keeps state fp32; only
// the O(1e-3) correction term passes through f16).
// vsite[b*vstride + p] gives the 2-vector v for this site.
// ---------------------------------------------------------------------------
__device__ __forceinline__ void gemm_env_update(float* env, const half_t* Bt,
                                                const float* __restrict__ vsite,
                                                int vstride, bool residual) {
  const int tid   = threadIdx.x;
  const int wave  = tid >> 5;
  const int lane  = tid & 31;
  const int lhalf = lane >> 4;   // K-half select (ISA A/B lane layout)
  const int ln    = lane & 15;   // M (for A) / N (for B) within tile

  const int   arow = wave * 16 + ln;              // batch row this lane feeds
  const float av0  = vsite[arow * vstride + 0];
  const float av1  = vsite[arow * vstride + 1];
  const float* erow = env + arow * ENVS;

  v8f c[4];
#pragma unroll
  for (int nt = 0; nt < 4; ++nt)
#pragma unroll
    for (int r = 0; r < 8; ++r)
      c[nt][r] = residual ? env[(wave * 16 + r + 8 * lhalf) * ENVS + nt * 16 + ln]
                          : 0.0f;

#pragma unroll
  for (int kt = 0; kt < 4; ++kt) {
    const int kb = kt * 32 + lhalf * 8;   // even start half-index
    // A fragment: halves kb..kb+7 (vgpr 0-3) and kb+16..kb+23 (vgpr 4-7);
    // k = (d<<1)|p  ->  A[b][k] = env[b][d] * v[b][p]
    v16h a;
#pragma unroll
    for (int i = 0; i < 4; ++i) {
      const float e0 = erow[(kb >> 1) + i];
      const float e1 = erow[(kb >> 1) + 8 + i];
      a[2 * i]         = (half_t)(e0 * av0);
      a[2 * i + 1]     = (half_t)(e0 * av1);
      a[8 + 2 * i]     = (half_t)(e1 * av0);
      a[8 + 2 * i + 1] = (half_t)(e1 * av1);
    }
#pragma unroll
    for (int nt = 0; nt < 4; ++nt) {
      const half_t* br = Bt + (nt * 16 + ln) * BTS + kb;
      v16h bf;
#pragma unroll
      for (int i = 0; i < 8; ++i) { bf[i] = br[i]; bf[8 + i] = br[16 + i]; }
      c[nt] = __builtin_amdgcn_wmma_f32_16x16x32_f16(
          false, a, false, bf, (short)0, c[nt], false, false);
    }
  }

  __syncthreads();                       // all env/A reads done before overwrite
#pragma unroll
  for (int nt = 0; nt < 4; ++nt)
#pragma unroll
    for (int r = 0; r < 8; ++r)
      env[(wave * 16 + r + 8 * lhalf) * ENVS + nt * 16 + ln] = c[nt][r];
  __syncthreads();
}

// ---------------------------------------------------------------------------
// Persistent scan over n sites (spacing 9). Block 0 = left env, block 1 = right.
// Records env *before* each output site into Lbuf/Rbuf[j] (B x 64).
// ---------------------------------------------------------------------------
__global__ __launch_bounds__(256) void smpo_scan(
    const float* __restrict__ v, const float* __restrict__ Wmid,
    const float* __restrict__ Wout, float* __restrict__ Lbuf,
    float* __restrict__ Rbuf, int n, int J) {
  (void)J;
  __shared__ float  env[128 * ENVS];
  __shared__ half_t Bt[64 * BTS];
  const int  tid   = threadIdx.x;
  const bool right = (blockIdx.x != 0);

  for (int q = tid; q < 128 * ENVS; q += BDIM) env[q] = 0.f;
  __syncthreads();
  if (tid < 128) env[tid * ENVS] = 1.f;          // bond-1 boundary
  __syncthreads();

  for (int step = 0; step < n; ++step) {
    const int  s       = right ? (n - 1 - step) : step;
    const int  j       = s / 9;
    const int  t       = s % 9;
    const bool outSite = (t == 0);

    if (outSite) {
      float* rec = (right ? Rbuf : Lbuf) + (size_t)j * 8192;
      for (int q = tid; q < 8192; q += BDIM)
        rec[q] = env[(q >> 6) * ENVS + (q & 63)];
      fill_Bt_out(Bt, Wout + (size_t)j * 73728, right);
    } else {
      fill_Bt_mid(Bt, Wmid + ((size_t)j * 8 + (t - 1)) * 8192, right);
    }

    // Prefetch next site's weights into cache (global_prefetch_b8 path).
    const int s2 = right ? (s - 1) : (s + 1);
    if (s2 >= 0 && s2 < n) {
      const int j2 = s2 / 9, t2 = s2 % 9;
      const float* wn = (t2 == 0) ? (Wout + (size_t)j2 * 73728)
                                  : (Wmid + ((size_t)j2 * 8 + (t2 - 1)) * 8192);
      __builtin_prefetch(wn + (size_t)tid * 32, 0, 1);
    }
    __syncthreads();

    gemm_env_update(env, Bt, v + (size_t)s * 2, n * 2, /*residual=*/!outSite);
  }
}

// ---------------------------------------------------------------------------
// Readout: h[b,j,f] = sum_{d,e,p} L[j,b,d] W[j,d,e,p,f] v[b,site j*9,p] R[j,b,e]
// One block per j; weight reads are wave-uniform (hardware broadcast).
// ---------------------------------------------------------------------------
__global__ __launch_bounds__(256) void smpo_readout(
    const float* __restrict__ Lbuf, const float* __restrict__ Rbuf,
    const float* __restrict__ Wout, const float* __restrict__ v,
    float* __restrict__ h, int n, int J) {
  __shared__ float Rs[128 * 65];
  const int j = blockIdx.x;
  for (int q = threadIdx.x; q < 8192; q += 256)
    Rs[(q >> 6) * 65 + (q & 63)] = Rbuf[(size_t)j * 8192 + q];
  __syncthreads();

  const int b  = threadIdx.x & 127;
  const int fh = threadIdx.x >> 7;      // uniform per wave
  const int f0 = fh ? 5 : 0;
  const int nf = fh ? 4 : 5;
  const float v0 = v[((size_t)b * n + (size_t)j * 9) * 2];
  const float v1 = v[((size_t)b * n + (size_t)j * 9) * 2 + 1];
  const float* Lrow = Lbuf + (size_t)j * 8192 + b * 64;
  const float* W    = Wout + (size_t)j * 73728;

  float acc[5] = {0.f, 0.f, 0.f, 0.f, 0.f};
  for (int d = 0; d < 64; ++d) {
    const float  ld = Lrow[d];
    const float* wd = W + d * 1152;
    for (int e = 0; e < 64; ++e) {
      const float  g  = ld * Rs[b * 65 + e];
      const float* wp = wd + e * 18 + f0;
      for (int f = 0; f < nf; ++f) acc[f] += g * (v0 * wp[f] + v1 * wp[9 + f]);
    }
  }
  for (int f = 0; f < nf; ++f)
    h[((size_t)b * J + j) * 9 + f0 + f] = acc[f];
}

// ---------------------------------------------------------------------------
// Patch embed: 3x3 patches -> normalized (cos,sin) mean.
// ---------------------------------------------------------------------------
__global__ void embed_patches(const float* __restrict__ img, float* __restrict__ v,
                              int B, int imgH, int ppr) {
  const int total = B * ppr * ppr;
  const int idx = blockIdx.x * blockDim.x + threadIdx.x;
  if (idx >= total) return;
  const int b  = idx / (ppr * ppr);
  const int pi = idx % (ppr * ppr);
  const int ph = pi / ppr, pw = pi % ppr;
  float cs = 0.f, sn = 0.f;
  for (int a = 0; a < 3; ++a)
    for (int c2 = 0; c2 < 3; ++c2) {
      const float x = img[(size_t)b * imgH * imgH + (size_t)(ph * 3 + a) * imgH +
                          (pw * 3 + c2)];
      float sv, cv;
      sincosf(1.5707963267948966f * x, &sv, &cv);
      cs += cv; sn += sv;
    }
  cs *= (1.f / 9.f); sn *= (1.f / 9.f);
  const float inv = rsqrtf(cs * cs + sn * sn);
  v[(size_t)idx * 2]     = cs * inv;
  v[(size_t)idx * 2 + 1] = sn * inv;
}

__global__ void embed_cos_sin(const float* __restrict__ x, float* __restrict__ v,
                              int total) {
  const int i = blockIdx.x * blockDim.x + threadIdx.x;
  if (i >= total) return;
  float sv, cv;
  sincosf(1.5707963267948966f * x[i], &sv, &cv);
  v[(size_t)i * 2]     = cv;
  v[(size_t)i * 2 + 1] = sv;
}

// ---------------------------------------------------------------------------
// Final SMPO (single output leg): one block per output index o; same WMMA
// scan over 80 mid sites; out[b,o] = u[b, e=0].
// ---------------------------------------------------------------------------
__global__ __launch_bounds__(256) void final_scan(
    const float* __restrict__ vF, const float* __restrict__ A0f,
    const float* __restrict__ WmidF, float* __restrict__ out) {
  __shared__ float  env[128 * ENVS];
  __shared__ half_t Bt[64 * BTS];
  const int tid = threadIdx.x;
  const int o   = blockIdx.x;

  // u[b,e] = v[b,0,0]*A0f[0,e,0,o] + v[b,0,1]*A0f[0,e,1,o]
  for (int q = tid; q < 8192; q += BDIM) {
    const int b = q >> 6, e = q & 63;
    env[b * ENVS + e] = vF[(size_t)b * 162] * A0f[e * 20 + o] +
                        vF[(size_t)b * 162 + 1] * A0f[e * 20 + 10 + o];
  }
  __syncthreads();

  for (int s = 1; s < 81; ++s) {
    fill_Bt_mid(Bt, WmidF + (size_t)(s - 1) * 8192, /*right=*/false);
    if (s < 80) __builtin_prefetch(WmidF + (size_t)s * 8192 + (size_t)tid * 32, 0, 1);
    __syncthreads();
    gemm_env_update(env, Bt, vF + (size_t)s * 2, 162, /*residual=*/true);
  }
  if (tid < 128) out[tid * 10 + o] = env[tid * ENVS];
}

// ---------------------------------------------------------------------------
extern "C" void kernel_launch(void* const* d_in, const int* in_sizes, int n_in,
                              void* d_out, int out_size, void* d_ws, size_t ws_size,
                              hipStream_t stream) {
  (void)in_sizes; (void)n_in; (void)out_size; (void)ws_size;
  const float* x     = (const float*)d_in[0];  // (128,81,81,1)
  const float* Wout0 = (const float*)d_in[1];  // (81,64,64,2,9)
  const float* Wmid0 = (const float*)d_in[2];  // (81,8,64,64,2)
  const float* Wout1 = (const float*)d_in[3];  // (9,64,64,2,9)
  const float* Wmid1 = (const float*)d_in[4];  // (9,8,64,64,2)
  const float* A0f   = (const float*)d_in[5];  // (64,64,2,10)
  const float* WmidF = (const float*)d_in[6];  // (80,64,64,2)
  float* out = (float*)d_out;                  // (128,10)

  float* ws = (float*)d_ws;
  size_t o = 0;
  float* v0 = ws + o; o += (size_t)128 * 729 * 2;
  float* L0 = ws + o; o += (size_t)81 * 8192;
  float* R0 = ws + o; o += (size_t)81 * 8192;
  float* h1 = ws + o; o += (size_t)128 * 729;
  float* v1 = ws + o; o += (size_t)128 * 81 * 2;
  float* L1 = ws + o; o += (size_t)9 * 8192;
  float* R1 = ws + o; o += (size_t)9 * 8192;
  float* h2 = ws + o; o += (size_t)128 * 81;
  float* vF = ws + o; o += (size_t)128 * 81 * 2;

  embed_patches<<<(128 * 729 + 255) / 256, 256, 0, stream>>>(x, v0, 128, 81, 27);
  smpo_scan<<<2, 256, 0, stream>>>(v0, Wmid0, Wout0, L0, R0, 729, 81);
  smpo_readout<<<81, 256, 0, stream>>>(L0, R0, Wout0, v0, h1, 729, 81);
  embed_patches<<<(128 * 81 + 255) / 256, 256, 0, stream>>>(h1, v1, 128, 27, 9);
  smpo_scan<<<2, 256, 0, stream>>>(v1, Wmid1, Wout1, L1, R1, 81, 9);
  smpo_readout<<<9, 256, 0, stream>>>(L1, R1, Wout1, v1, h2, 81, 9);
  embed_cos_sin<<<(128 * 81 + 255) / 256, 256, 0, stream>>>(h2, vF, 128 * 81);
  final_scan<<<10, 256, 0, stream>>>(vF, A0f, WmidF, out);
}
